// DeformableTransformerEncoder_11579231830641
// MI455X (gfx1250) — matrix operand order repaired and checked by reading, not compiled, not run on hardware
//
#include <hip/hip_runtime.h>
#include <hip/hip_bf16.h>
#include <math.h>

// ---------------- problem constants ----------------
#define B_       2
#define C_       256
#define NH_      8
#define NL_      4
#define NP_      4
#define HD_      32
#define DFF_     1024
#define NLAYERS_ 6
#define NDEC_    6
#define NQ_      300
#define N_TOTAL_ 11253
#define TOPK_    5626

__constant__ int c_HW[4]  = {92, 46, 23, 12};          // square levels: H == W
__constant__ int c_LSI[4] = {0, 8464, 10580, 11109};   // level start indices

typedef __attribute__((ext_vector_type(16))) _Float16 v16h;
typedef __attribute__((ext_vector_type(8)))  _Float16 v8h;
typedef __attribute__((ext_vector_type(4)))  _Float16 v4h;
typedef __attribute__((ext_vector_type(8)))  float    v8f;
typedef __attribute__((ext_vector_type(4)))  int      v4i;
// address-space-qualified int4 pointers for the async-LDS builtin
typedef __attribute__((address_space(1))) v4i g_v4i;
typedef __attribute__((address_space(3))) v4i l_v4i;

#if __has_builtin(__builtin_amdgcn_global_load_async_to_lds_b128)
#define HAVE_ASYNC_LDS 1
#else
#define HAVE_ASYNC_LDS 0
#endif

// =====================================================================
// WMMA GEMM:  C[M,N] = A[M,K](f32) x Wt[N,K](f16, pre-transposed)
// block tile 64x32, 8 waves (4x2 of 16x16), K-step 64 (2 WMMA / wave / step)
// Double-buffered LDS: async B-copy + A global loads for tile k+1 overlap
// the WMMAs on tile k.
// FLAGS: bit0 = +bias[col], bit1 = relu, bit2 = +resid[row,col]
// =====================================================================
#define BM 64
#define BN 32
#define BK 64

template <int FLAGS>
__global__ void __launch_bounds__(256)
gemm_f16wmma(const float* __restrict__ A, const _Float16* __restrict__ Bt,
             const float* __restrict__ bias, const float* __restrict__ resid,
             float* __restrict__ Cm, int M, int N, int K)
{
    __shared__ _Float16 As[2][BM][BK] __attribute__((aligned(128)));   // 16 KB
    __shared__ _Float16 Bs[2][BN][BK] __attribute__((aligned(128)));   //  8 KB

    const int tid  = threadIdx.x;
    const int bm   = blockIdx.x * BM;
    const int bn   = blockIdx.y * BN;
    const int wave = tid >> 5;
    const int lane = tid & 31;
    const int wm   = (wave >> 1) * 16;   // 0,16,32,48
    const int wn   = (wave & 1) * 16;    // 0,16
    const int l16  = lane & 15;
    const int hi   = lane >> 4;          // 0 or 1

    // B-tile copy mapping: 32 rows x 8 chunks of 16B (f16 [N,K] rows)
    const int brow = tid >> 3;           // 0..31
    const int bch  = tid & 7;            // 0..7

    // issue async copy of one B tile into LDS buffer `buf`
    auto issueB = [&](int k0, int buf) {
        const _Float16* g = Bt + (size_t)(bn + brow) * K + k0 + bch * 8;
        _Float16* l = &Bs[buf][brow][bch * 8];
#if HAVE_ASYNC_LDS
        __builtin_amdgcn_global_load_async_to_lds_b128((g_v4i*)g, (l_v4i*)l, 0, 0);
#else
        *(v8h*)l = *(const v8h*)g;
#endif
    };
    // load one A tile (64x64 f32) into registers, rows clamped (no divergence)
    auto loadA = [&](int k0, float4* r) {
        #pragma unroll
        for (int i = 0; i < 4; ++i) {
            int idx = (tid + i * 256) * 4;
            int m  = idx >> 6;
            int kk = idx & 63;
            int gm = bm + m;
            gm = gm < M ? gm : M - 1;
            r[i] = *(const float4*)&A[(size_t)gm * K + k0 + kk];
        }
    };
    // convert + store A registers into LDS buffer `buf`
    auto storeA = [&](const float4* r, int buf) {
        #pragma unroll
        for (int i = 0; i < 4; ++i) {
            int idx = (tid + i * 256) * 4;
            int m  = idx >> 6;
            int kk = idx & 63;
            v4h h4; h4[0] = (_Float16)r[i].x; h4[1] = (_Float16)r[i].y;
                    h4[2] = (_Float16)r[i].z; h4[3] = (_Float16)r[i].w;
            *(v4h*)&As[buf][m][kk] = h4;
        }
    };

    v8f acc = {};
    float4 areg[4];

    // ---- prologue: stage tile 0 ----
    issueB(0, 0);
    loadA(0, areg);
    storeA(areg, 0);
#if HAVE_ASYNC_LDS
    asm volatile("s_wait_asynccnt 0" ::: "memory");
#endif
    __syncthreads();

    const int nk = K / BK;
    for (int kt = 0; kt < nk; ++kt) {
        const int cur = kt & 1;
        const int nxt = cur ^ 1;
        const bool more = (kt + 1) < nk;

        // start fetching tile k+1 (overlaps the WMMAs below)
        if (more) {
            issueB((kt + 1) * BK, nxt);
            loadA((kt + 1) * BK, areg);
        }

        // ---- two 16x16x32 WMMAs on tile k ----
        #pragma unroll
        for (int ks = 0; ks < BK; ks += 32) {
            // A fragment: contiguous 16-bit A layout (matches B)
            // lane<16 : K = ks+0..15; lane>=16 : K = ks+16..31
            const int am = wm + l16;
            const v16h afrag = *(const v16h*)&As[cur][am][ks + hi * 16];
            // B fragment: lane<16 -> K=ks..ks+15, lane>=16 -> K=ks+16..ks+31
            const v16h bfrag = *(const v16h*)&Bs[cur][wn + l16][ks + hi * 16];

            acc = __builtin_amdgcn_wmma_f32_16x16x32_f16(
                      false, afrag, false, bfrag, (short)0, acc, false, false);
        }

        if (more) storeA(areg, nxt);
#if HAVE_ASYNC_LDS
        asm volatile("s_wait_asynccnt 0" ::: "memory");
#endif
        __syncthreads();
    }

    // ---- epilogue: lane<16: N=lane, M=r ; lane>=16: M=8+r ----
    const int col  = bn + wn + l16;
    const int row0 = bm + wm + hi * 8;
    const float badd = (FLAGS & 1) ? bias[col] : 0.0f;
    if (bm + BM <= M) {                      // full tile: no guards
        #pragma unroll
        for (int r = 0; r < 8; ++r) {
            int row = row0 + r;
            float v = acc[r] + badd;
            if (FLAGS & 2) v = v > 0.0f ? v : 0.0f;
            if (FLAGS & 4) v += resid[(size_t)row * N + col];
            Cm[(size_t)row * N + col] = v;
        }
    } else {
        #pragma unroll
        for (int r = 0; r < 8; ++r) {
            int row = row0 + r;
            if (row < M) {
                float v = acc[r] + badd;
                if (FLAGS & 2) v = v > 0.0f ? v : 0.0f;
                if (FLAGS & 4) v += resid[(size_t)row * N + col];
                Cm[(size_t)row * N + col] = v;
            }
        }
    }
}

// ---- weight convert + transpose: Wt[n*K + k] = (f16) W[k*N + n] ----
__global__ void convw_kernel(const float* __restrict__ W, _Float16* __restrict__ Wt,
                             int K, int N)
{
    int gid = blockIdx.x * blockDim.x + threadIdx.x;
    if (gid >= K * N) return;
    int k = gid % K, n = gid / K;
    Wt[gid] = (_Float16)W[(size_t)k * N + n];
}

// =====================================================================
// utility kernels
// =====================================================================
__global__ void copy_kernel(const float* __restrict__ s, float* __restrict__ d, size_t n) {
    size_t i = (size_t)blockIdx.x * blockDim.x + threadIdx.x;
    if (i < n) d[i] = s[i];
}
__global__ void zero_kernel(float* __restrict__ d, size_t n) {
    size_t i = (size_t)blockIdx.x * blockDim.x + threadIdx.x;
    if (i < n) d[i] = 0.0f;
}
__global__ void add_kernel(const float* __restrict__ a, const float* __restrict__ b,
                           float* __restrict__ d, size_t n) {
    size_t i = (size_t)blockIdx.x * blockDim.x + threadIdx.x;
    if (i < n) d[i] = a[i] + b[i];
}

// ---- scatter decoder attention maps to flat saliency grid ----
__global__ void flat_grid_kernel(const float* __restrict__ dec_loc,
                                 const float* __restrict__ dec_w,
                                 float* __restrict__ flat)
{
    int gid = blockIdx.x * blockDim.x + threadIdx.x;
    const int total = B_ * NDEC_ * NQ_ * NH_ * NL_ * NP_;
    if (gid >= total) return;
    int p = gid % NP_;  int t = gid / NP_;
    int l = t % NL_;    t /= NL_;
    int b = (t / (NH_ * NQ_ * NDEC_));
    (void)p;

    size_t base = (size_t)gid;
    float sx = dec_loc[base * 2 + 0];
    float sy = dec_loc[base * 2 + 1];
    float w  = dec_w[base];

    int HW = c_HW[l], lsi = c_LSI[l];
    float cx = sx * HW, cy = sy * HW;
    int lx = (int)floorf(cx), ly = (int)floorf(cy);
    float fx = cx - (float)lx, fy = cy - (float)ly;

    float* fb = flat + (size_t)b * N_TOTAL_;
    #pragma unroll
    for (int c4 = 0; c4 < 4; ++c4) {
        int dx = c4 >> 1, dy = c4 & 1;
        int xi = lx + dx, yi = ly + dy;
        float m = (dx ? fx : 1.0f - fx) * (dy ? fy : 1.0f - fy);
        if (xi >= 0 && xi < HW && yi >= 0 && yi < HW) {
            atomicAdd(&fb[yi * HW + xi + lsi], w * m);
        }
    }
}

// ---- deterministic top-k via rank counting (desc value, asc index ties) ----
__global__ void __launch_bounds__(256)
topk_kernel(const float* __restrict__ flat, int* __restrict__ inds)
{
    int b = blockIdx.y;
    int i = blockIdx.x * 256 + threadIdx.x;
    const float* f = flat + (size_t)b * N_TOTAL_;
    float vi = (i < N_TOTAL_) ? f[i] : -1e30f;
    int rank = 0;
    __shared__ float tile[256];
    for (int j0 = 0; j0 < N_TOTAL_; j0 += 256) {
        int j = j0 + threadIdx.x;
        tile[threadIdx.x] = (j < N_TOTAL_) ? f[j] : -1e30f;
        __syncthreads();
        int lim = (N_TOTAL_ - j0 < 256) ? (N_TOTAL_ - j0) : 256;
        for (int jj = 0; jj < lim; ++jj) {
            float vj = tile[jj];
            int gj = j0 + jj;
            if (vj > vi || (vj == vi && gj < i)) rank++;
        }
        __syncthreads();
    }
    if (i < N_TOTAL_ && rank < TOPK_) inds[(size_t)b * TOPK_ + rank] = i;
}

// ---- gather tgt / pos_q and compute per-query reference points ----
__global__ void gather_kernel(const float* __restrict__ src, const float* __restrict__ pos,
                              const int* __restrict__ inds, const float* __restrict__ vr,
                              float* __restrict__ tgt, float* __restrict__ posq,
                              float* __restrict__ ref)
{
    size_t gid = (size_t)blockIdx.x * blockDim.x + threadIdx.x;
    const size_t n = (size_t)B_ * TOPK_ * C_;
    if (gid >= n) return;
    int c = gid % C_;
    size_t t = gid / C_;
    int tq = t % TOPK_;
    int b  = t / TOPK_;
    int idx = inds[(size_t)b * TOPK_ + tq];
    size_t si = ((size_t)b * N_TOTAL_ + idx) * C_ + c;
    tgt[gid]  = src[si];
    posq[gid] = pos[si];
    if (c < NL_ * 2) {
        int lvl = (idx >= 11109) ? 3 : (idx >= 10580) ? 2 : (idx >= 8464) ? 1 : 0;
        int HW = c_HW[lvl];
        int nn = idx - c_LSI[lvl];
        int row = nn / HW, colp = nn % HW;
        float vrx = vr[((size_t)b * NL_ + lvl) * 2 + 0];
        float vry = vr[((size_t)b * NL_ + lvl) * 2 + 1];
        float rx = ((float)colp + 0.5f) / (vrx * (float)HW);
        float ry = ((float)row  + 0.5f) / (vry * (float)HW);
        int ol = c >> 1, xy = c & 1;
        float s = vr[((size_t)b * NL_ + ol) * 2 + xy];
        ref[((size_t)b * TOPK_ + tq) * (NL_ * 2) + c] = (xy == 0 ? rx : ry) * s;
    }
}

// ---- per-(query,head): softmax over 16 attn logits + sampling locations ----
__global__ void softmax_loc_kernel(const float* __restrict__ attnraw,
                                   const float* __restrict__ offb,
                                   const float* __restrict__ ref,
                                   float* __restrict__ aw, float* __restrict__ samp)
{
    int gid = blockIdx.x * blockDim.x + threadIdx.x;
    const int total = B_ * TOPK_ * NH_;
    if (gid >= total) return;
    int h = gid % NH_;
    size_t bt = gid / NH_;

    const float* ar = attnraw + bt * (NH_ * NL_ * NP_) + h * 16;
    float mx = ar[0];
    #pragma unroll
    for (int i = 1; i < 16; ++i) mx = fmaxf(mx, ar[i]);
    float e[16]; float s = 0.0f;
    #pragma unroll
    for (int i = 0; i < 16; ++i) { e[i] = expf(ar[i] - mx); s += e[i]; }
    float invs = 1.0f / s;
    float* awp = aw + (size_t)gid * 16;
    #pragma unroll
    for (int i = 0; i < 16; ++i) awp[i] = e[i] * invs;

    const float* off = offb + bt * C_ + h * (NL_ * NP_ * 2);
    const float* rf  = ref + bt * (NL_ * 2);
    float* sp = samp + (size_t)gid * (NL_ * NP_ * 2);
    #pragma unroll
    for (int l = 0; l < NL_; ++l) {
        float invw = 1.0f / (float)c_HW[l];
        #pragma unroll
        for (int p = 0; p < NP_; ++p) {
            int o = (l * NP_ + p) * 2;
            sp[o + 0] = rf[l * 2 + 0] + off[o + 0] * invw;
            sp[o + 1] = rf[l * 2 + 1] + off[o + 1] * invw;
        }
    }
}

// ---- multi-scale deformable attention: one wave per (query, head),
//      lane = head-dim channel (HD_ == 32 == wave width) ----
__global__ void __launch_bounds__(256)
msda_kernel(const float* __restrict__ value, const float* __restrict__ samp,
            const float* __restrict__ aw, float* __restrict__ out)
{
    int gtid = blockIdx.x * blockDim.x + threadIdx.x;
    int wave = gtid >> 5;
    int lane = gtid & 31;
    const int total = B_ * TOPK_ * NH_;
    if (wave >= total) return;
    int h = wave % NH_;
    size_t bt = wave / NH_;
    int b = (int)(bt / TOPK_);

    const float* loc = samp + (size_t)wave * (NL_ * NP_ * 2);
    const float* awp = aw + (size_t)wave * (NL_ * NP_);
    float acc = 0.0f;
    int start = 0;
    #pragma unroll
    for (int l = 0; l < NL_; ++l) {
        int HW = c_HW[l];
        const float* v = value + ((size_t)b * N_TOTAL_ + start) * C_ + h * HD_;
        #pragma unroll
        for (int p = 0; p < NP_; ++p) {
            float x = loc[(l * NP_ + p) * 2 + 0] * (float)HW - 0.5f;
            float y = loc[(l * NP_ + p) * 2 + 1] * (float)HW - 0.5f;
            float x0 = floorf(x), y0 = floorf(y);
            float fx = x - x0, fy = y - y0;
            int x0i = (int)x0, y0i = (int)y0;
            float a = awp[l * NP_ + p];
            #pragma unroll
            for (int c4 = 0; c4 < 4; ++c4) {
                int dx = c4 & 1, dy = c4 >> 1;
                int xi = x0i + dx, yi = y0i + dy;
                float w = (dx ? fx : 1.0f - fx) * (dy ? fy : 1.0f - fy);
                bool valid = (xi >= 0) & (xi < HW) & (yi >= 0) & (yi < HW);
                int xc = xi < 0 ? 0 : (xi > HW - 1 ? HW - 1 : xi);
                int yc = yi < 0 ? 0 : (yi > HW - 1 ? HW - 1 : yi);
                float g = v[(size_t)(yc * HW + xc) * C_ + lane];
                acc += g * (valid ? w * a : 0.0f);
            }
        }
        start += HW * HW;
    }
    out[bt * C_ + h * HD_ + lane] = acc;
}

// ---- layernorm over C=256, one row per block ----
__global__ void __launch_bounds__(256)
ln_kernel(const float* __restrict__ x, const float* __restrict__ g,
          const float* __restrict__ bta, float* __restrict__ y, int rows)
{
    int row = blockIdx.x;
    if (row >= rows) return;
    int t = threadIdx.x;
    float v = x[(size_t)row * C_ + t];
    __shared__ float red[256];
    red[t] = v; __syncthreads();
    for (int s = 128; s > 0; s >>= 1) { if (t < s) red[t] += red[t + s]; __syncthreads(); }
    float mu = red[0] / (float)C_;
    __syncthreads();
    float d = v - mu;
    red[t] = d * d; __syncthreads();
    for (int s = 128; s > 0; s >>= 1) { if (t < s) red[t] += red[t + s]; __syncthreads(); }
    float var = red[0] / (float)C_;
    float r = rsqrtf(var + 1e-5f);
    y[(size_t)row * C_ + t] = d * r * g[t] + bta[t];
}

// ---- scatter updated queries back into the dense output ----
__global__ void scatter_kernel(const float* __restrict__ tgt, const int* __restrict__ inds,
                               float* __restrict__ out)
{
    size_t gid = (size_t)blockIdx.x * blockDim.x + threadIdx.x;
    const size_t n = (size_t)B_ * TOPK_ * C_;
    if (gid >= n) return;
    int c = gid % C_;
    size_t t = gid / C_;
    int tq = t % TOPK_;
    int b  = t / TOPK_;
    int idx = inds[(size_t)b * TOPK_ + tq];
    out[((size_t)b * N_TOTAL_ + idx) * C_ + c] = tgt[gid];
}

// =====================================================================
// host
// =====================================================================
static void launch_gemm(const float* A, const _Float16* Bt, const float* bias,
                        const float* resid, float* Cm, int M, int N, int K,
                        int flags, hipStream_t s)
{
    dim3 grid((unsigned)((M + BM - 1) / BM), (unsigned)(N / BN));
    if (flags == (1 | 2))
        gemm_f16wmma<1 | 2><<<grid, 256, 0, s>>>(A, Bt, bias, resid, Cm, M, N, K);
    else if (flags == (1 | 4))
        gemm_f16wmma<1 | 4><<<grid, 256, 0, s>>>(A, Bt, bias, resid, Cm, M, N, K);
    else
        gemm_f16wmma<1><<<grid, 256, 0, s>>>(A, Bt, bias, resid, Cm, M, N, K);
}

static void conv_w(const float* W, _Float16* Wt, int K, int N, hipStream_t s)
{
    int tot = K * N;
    convw_kernel<<<(tot + 255) / 256, 256, 0, s>>>(W, Wt, K, N);
}

extern "C" void kernel_launch(void* const* d_in, const int* in_sizes, int n_in,
                              void* d_out, int out_size, void* d_ws, size_t ws_size,
                              hipStream_t stream)
{
    (void)in_sizes; (void)n_in; (void)out_size; (void)ws_size;

    const float* src      = (const float*)d_in[0];
    const float* vr       = (const float*)d_in[3];
    const float* pos      = (const float*)d_in[4];
    const float* dec_loc  = (const float*)d_in[6];
    const float* dec_w    = (const float*)d_in[7];
    const float* W_off    = (const float*)d_in[8];
    const float* b_off    = (const float*)d_in[9];
    const float* W_attn   = (const float*)d_in[10];
    const float* b_attn   = (const float*)d_in[11];
    const float* W_v      = (const float*)d_in[12];
    const float* b_v      = (const float*)d_in[13];
    const float* W_o      = (const float*)d_in[14];
    const float* b_o      = (const float*)d_in[15];
    const float* ln1_g    = (const float*)d_in[16];
    const float* ln1_b    = (const float*)d_in[17];
    const float* W_ff1    = (const float*)d_in[18];
    const float* b_ff1    = (const float*)d_in[19];
    const float* W_ff2    = (const float*)d_in[20];
    const float* b_ff2    = (const float*)d_in[21];
    const float* ln2_g    = (const float*)d_in[22];
    const float* ln2_b    = (const float*)d_in[23];

    float* out_output = (float*)d_out;                                   // (B,1,N,C)
    float* out_samp   = out_output + (size_t)B_ * N_TOTAL_ * C_;         // (B,TOPK,NH,NL,NP,2)

    float* ws = (float*)d_ws;
    size_t cur = 0;
    auto alloc = [&](size_t nfl) { float* p = ws + cur; cur += (nfl + 63) & ~(size_t)63; return p; };
    float* flat    = alloc((size_t)B_ * N_TOTAL_);
    int*   inds    = (int*)alloc((size_t)B_ * TOPK_);
    float* ref     = alloc((size_t)B_ * TOPK_ * NL_ * 2);
    float* tgt     = alloc((size_t)B_ * TOPK_ * C_);
    float* posq    = alloc((size_t)B_ * TOPK_ * C_);
    float* qbuf    = alloc((size_t)B_ * TOPK_ * C_);
    float* value   = alloc((size_t)B_ * N_TOTAL_ * C_);
    float* offb    = alloc((size_t)B_ * TOPK_ * C_);
    float* attnraw = alloc((size_t)B_ * TOPK_ * NH_ * NL_ * NP_);
    float* awbuf   = alloc((size_t)B_ * TOPK_ * NH_ * NL_ * NP_);
    float* attn_in = alloc((size_t)B_ * TOPK_ * C_);
    float* tmp     = alloc((size_t)B_ * TOPK_ * C_);
    float* h1      = alloc((size_t)B_ * TOPK_ * DFF_);
    // f16 pre-transposed weights: per layer 3*65536 + 32768 + 2*262144 halves
    const size_t WPL = 753664;                         // halves per layer
    _Float16* w16 = (_Float16*)alloc(WPL * NLAYERS_ / 2 + 64);

    const size_t nOut = (size_t)B_ * N_TOTAL_ * C_;
    const size_t nQ   = (size_t)B_ * TOPK_ * C_;
    const int    Mq   = B_ * TOPK_;
    const int    Mv   = B_ * N_TOTAL_;

    // ---- convert + transpose all weights to f16 [N,K] (L2-resident) ----
    for (int lid = 0; lid < NLAYERS_; ++lid) {
        _Float16* wl = w16 + (size_t)lid * WPL;
        conv_w(W_v   + (size_t)lid * C_ * C_,    wl,          C_,  C_,   stream);
        conv_w(W_off + (size_t)lid * C_ * C_,    wl + 65536,  C_,  C_,   stream);
        conv_w(W_attn+ (size_t)lid * C_ * 128,   wl + 131072, C_,  128,  stream);
        conv_w(W_o   + (size_t)lid * C_ * C_,    wl + 163840, C_,  C_,   stream);
        conv_w(W_ff1 + (size_t)lid * C_ * DFF_,  wl + 229376, C_,  DFF_, stream);
        conv_w(W_ff2 + (size_t)lid * DFF_ * C_,  wl + 491520, DFF_, C_,  stream);
    }

    // output starts as src; scatter-updated after each layer
    copy_kernel<<<(unsigned)((nOut + 255) / 256), 256, 0, stream>>>(src, out_output, nOut);

    // saliency -> deterministic top-k -> gather
    zero_kernel<<<(unsigned)((B_ * N_TOTAL_ + 255) / 256), 256, 0, stream>>>(flat, (size_t)B_ * N_TOTAL_);
    int nScat = B_ * NDEC_ * NQ_ * NH_ * NL_ * NP_;
    flat_grid_kernel<<<(nScat + 255) / 256, 256, 0, stream>>>(dec_loc, dec_w, flat);
    dim3 tkg((N_TOTAL_ + 255) / 256, B_);
    topk_kernel<<<tkg, 256, 0, stream>>>(flat, inds);
    gather_kernel<<<(unsigned)((nQ + 255) / 256), 256, 0, stream>>>(src, pos, inds, vr, tgt, posq, ref);

    for (int lid = 0; lid < NLAYERS_; ++lid) {
        _Float16* wl = w16 + (size_t)lid * WPL;
        const _Float16* wv16 = wl;
        const _Float16* wf16 = wl + 65536;
        const _Float16* wa16 = wl + 131072;
        const _Float16* wo16 = wl + 163840;
        const _Float16* w116 = wl + 229376;
        const _Float16* w216 = wl + 491520;

        add_kernel<<<(unsigned)((nQ + 255) / 256), 256, 0, stream>>>(tgt, posq, qbuf, nQ);

        // value = output @ W_v + b_v   (full sequence)
        launch_gemm(out_output, wv16, b_v + lid * C_, nullptr, value, Mv, C_, C_, 1, stream);
        // offsets & attention logits from q
        launch_gemm(qbuf, wf16, b_off + lid * C_, nullptr, offb, Mq, C_, C_, 1, stream);
        launch_gemm(qbuf, wa16, b_attn + lid * 128, nullptr, attnraw, Mq, 128, C_, 1, stream);

        int nSL = B_ * TOPK_ * NH_;
        softmax_loc_kernel<<<(nSL + 255) / 256, 256, 0, stream>>>(attnraw, offb, ref, awbuf, out_samp);
        msda_kernel<<<(unsigned)(((size_t)nSL * 32 + 255) / 256), 256, 0, stream>>>(
            value, out_samp, awbuf, attn_in);

        // out-proj + residual, LN1
        launch_gemm(attn_in, wo16, b_o + lid * C_, tgt, tmp, Mq, C_, C_, 1 | 4, stream);
        ln_kernel<<<Mq, 256, 0, stream>>>(tmp, ln1_g + lid * C_, ln1_b + lid * C_, tgt, Mq);
        // FFN: relu(tgt @ W1 + b1) @ W2 + b2 + tgt, LN2
        launch_gemm(tgt, w116, b_ff1 + lid * DFF_, nullptr, h1, Mq, DFF_, C_, 1 | 2, stream);
        launch_gemm(h1, w216, b_ff2 + lid * C_, tgt, tmp, Mq, C_, DFF_, 1 | 4, stream);
        ln_kernel<<<Mq, 256, 0, stream>>>(tmp, ln2_g + lid * C_, ln2_b + lid * C_, tgt, Mq);

        scatter_kernel<<<(unsigned)((nQ + 255) / 256), 256, 0, stream>>>(tgt, inds, out_output);
    }
    // out_samp already holds last layer's sampling locations (tuple output #2)
}
